// GATEncoder_11029476016727
// MI455X (gfx1250) — matrix operand order repaired and checked by reading, not compiled, not run on hardware
//
#include <hip/hip_runtime.h>
#include <hip/hip_bf16.h>
#include <stdint.h>

typedef __attribute__((ext_vector_type(16))) _Float16 v16h;
typedef __attribute__((ext_vector_type(8)))  float    v8f;

// ---------------- problem constants (from reference) ----------------
constexpr int NNODES = 50000;
constexpr int NEDGES = 1600000;
constexpr int NETOT  = NEDGES + NNODES;   // edges + self-loops = 1,650,000
constexpr int INCH   = 128;
constexpr int H1 = 4, C1 = 32, F1 = H1 * C1;   // 128
constexpr int F2 = 256;                         // layer-2: 1 head x 256
constexpr int KDIM = 128;                       // K of both GEMMs

// ---------------- small utility kernels ----------------
__global__ void k_zero_f4(float4* __restrict__ p, int n4) {
    int i = blockIdx.x * blockDim.x + threadIdx.x;
    if (i < n4) p[i] = make_float4(0.f, 0.f, 0.f, 0.f);
}

// Pack W[k][n] (f32, K=128 rows, Nout cols) into per-lane WMMA B-fragment image:
// Bp[((nt*4 + kstep)*32 + lane)*16 + e]  with  k = kstep*32 + (lane/16)*16 + e,
//                                              n = nt*16 + (lane%16)
__global__ void k_pack_B(const float* __restrict__ W, _Float16* __restrict__ Bp, int Nout) {
    int i = blockIdx.x * blockDim.x + threadIdx.x;      // over Nout*128
    if (i >= Nout * KDIM) return;
    const int e     = i & 15;
    const int lane  = (i >> 4) & 31;
    const int kstep = (i >> 9) & 3;                     // K/32 = 4
    const int nt    = i >> 11;
    const int k     = (kstep << 5) + ((lane >> 4) << 4) + e;
    const int n     = (nt << 4) + (lane & 15);
    Bp[i] = (_Float16)W[(size_t)k * Nout + n];
}

// ---------------- WMMA GEMM: C[M x Nout] = A[M x 128] * B[128 x Nout] ----------------
// A fp32 row-major, staged into LDS in per-lane A-fragment image; Bp pre-packed (above).
// Block = 256 threads = 8 waves; block owns 16 rows; each wave computes 16x16 tiles.
__global__ void k_gemm_wmma(const float* __restrict__ A,
                            const v16h* __restrict__ Bp,
                            float* __restrict__ Cout,
                            int Nout) {
    __shared__ v16h Atile[4 * 32];                      // [kstep][lane], 4 KB, 32B aligned

    const int tid  = threadIdx.x;
    const int row0 = blockIdx.x * 16;

    // Stage A-block (16 rows x 128 K) directly in WMMA A-fragment order.
    // A 16x32 layout (matches packed B): lane -> row M = lane%16, half = lane/16;
    // half-element e: K = half*16 + e
    _Float16* At = (_Float16*)Atile;
    for (int i = tid; i < 4 * 32 * 16; i += 256) {
        const int e     = i & 15;
        const int lane  = (i >> 4) & 31;
        const int kstep = i >> 9;
        const int m     = lane & 15;
        const int hf    = lane >> 4;
        const int v     = e >> 1, sl = e & 1;
        const int ka    = (hf << 4) + (v << 1) + sl;
        At[i] = (_Float16)A[(size_t)(row0 + m) * KDIM + (kstep << 5) + ka];
    }
    __syncthreads();

    const int wave = tid >> 5;
    const int lane = tid & 31;
    const int m    = lane & 15;
    const int hf   = lane >> 4;

    // A fragments are shared by every N-tile of this wave: hoist into registers.
    const v16h a0 = Atile[0 * 32 + lane];
    const v16h a1 = Atile[1 * 32 + lane];
    const v16h a2 = Atile[2 * 32 + lane];
    const v16h a3 = Atile[3 * 32 + lane];

    for (int nt = wave; nt < (Nout >> 4); nt += 8) {
        const v16h* __restrict__ bp = Bp + (size_t)nt * (4 * 32) + lane;
        v8f acc = {};
        acc = __builtin_amdgcn_wmma_f32_16x16x32_f16(false, a0, false, bp[0 * 32], (short)0, acc, false, false);
        acc = __builtin_amdgcn_wmma_f32_16x16x32_f16(false, a1, false, bp[1 * 32], (short)0, acc, false, false);
        acc = __builtin_amdgcn_wmma_f32_16x16x32_f16(false, a2, false, bp[2 * 32], (short)0, acc, false, false);
        acc = __builtin_amdgcn_wmma_f32_16x16x32_f16(false, a3, false, bp[3 * 32], (short)0, acc, false, false);
        // C layout: VGPR r -> row M = half*8 + r, col = lane%16
        const int ncol = (nt << 4) + m;
#pragma unroll
        for (int r = 0; r < 8; ++r) {
            Cout[(size_t)(row0 + (hf << 3) + r) * Nout + ncol] = acc[r];
        }
    }
}

// ---------------- per-node attention terms: al[n,h] = sum_c h[n,h,c]*a[h,c] ----------------
template <int H, int C>
__global__ void k_attn_terms(const float* __restrict__ hp,
                             const float* __restrict__ a_src,
                             const float* __restrict__ a_dst,
                             float* __restrict__ als,
                             float* __restrict__ ald) {
    int i = blockIdx.x * blockDim.x + threadIdx.x;   // over NNODES*H
    if (i >= NNODES * H) return;
    const int n = i / H, h = i - n * H;
    const float* row = hp + (size_t)n * (H * C) + h * C;
    const float* ws  = a_src + h * C;
    const float* wd  = a_dst + h * C;
    float ss = 0.f, sd = 0.f;
#pragma unroll 8
    for (int c = 0; c < C; ++c) {
        float v = row[c];
        ss += v * ws[c];
        sd += v * wd[c];
    }
    als[i] = ss;
    ald[i] = sd;
}

// ---------------- edge helpers ----------------
__device__ __forceinline__ void edge_sd(const int* __restrict__ ei, int e, int& s, int& d) {
    if (e < NEDGES) { s = ei[e]; d = ei[NEDGES + e]; }
    else            { s = e - NEDGES; d = s; }       // self-loops
}

__device__ __forceinline__ unsigned f32_sortable(float f) {
    unsigned b = __float_as_uint(f);
    return (b & 0x80000000u) ? ~b : (b | 0x80000000u);
}
__device__ __forceinline__ float sortable_f32(unsigned u) {
    return (u & 0x80000000u) ? __uint_as_float(u ^ 0x80000000u)
                             : __uint_as_float(~u);
}

// pass 1: per-edge LeakyReLU logits -> segment max over dst (atomicMax on sortable bits)
template <int H>
__global__ void k_edge_max(const int* __restrict__ ei,
                           const float* __restrict__ als,
                           const float* __restrict__ ald,
                           unsigned* __restrict__ mx) {
    int i = blockIdx.x * blockDim.x + threadIdx.x;   // over NETOT*H
    if (i >= NETOT * H) return;
    const int e = i / H, h = i - e * H;
    int s, d; edge_sd(ei, e, s, d);
    float lg = als[s * H + h] + ald[d * H + h];
    lg = lg > 0.f ? lg : 0.2f * lg;
    atomicMax(&mx[d * H + h], f32_sortable(lg));
}

// pass 2: e = exp(logit - max), store per-edge, segment-sum over dst
template <int H>
__global__ void k_edge_exp(const int* __restrict__ ei,
                           const float* __restrict__ als,
                           const float* __restrict__ ald,
                           const unsigned* __restrict__ mx,
                           float* __restrict__ eb,
                           float* __restrict__ ssum) {
    int i = blockIdx.x * blockDim.x + threadIdx.x;   // over NETOT*H
    if (i >= NETOT * H) return;
    const int e = i / H, h = i - e * H;
    int s, d; edge_sd(ei, e, s, d);
    float lg = als[s * H + h] + ald[d * H + h];
    lg = lg > 0.f ? lg : 0.2f * lg;
    float ev = __expf(lg - sortable_f32(mx[d * H + h]));
    eb[i] = ev;
    atomicAdd(&ssum[d * H + h], ev);
}

// pass 3: out[dst] += h[src] * alpha  (32 channels per thread, float4 gathers, L2 atomics)
template <int H, int C>
__global__ void k_edge_aggr(const int* __restrict__ ei,
                            const float* __restrict__ hp,
                            const float* __restrict__ eb,
                            const float* __restrict__ ssum,
                            float* __restrict__ agg) {
    constexpr int CH = C / 32;
    int i = blockIdx.x * blockDim.x + threadIdx.x;   // over NETOT*H*CH
    if (i >= NETOT * H * CH) return;
    const int e   = i / (H * CH);
    const int rem = i - e * (H * CH);
    const int h   = rem / CH;
    const int ch  = rem - h * CH;
    int s, d; edge_sd(ei, e, s, d);
    const float alpha = eb[e * H + h] / (ssum[d * H + h] + 1e-16f);
    const float4* __restrict__ srow =
        (const float4*)(hp + (size_t)s * (H * C) + h * C + ch * 32);
    float* __restrict__ drow = agg + (size_t)d * (H * C) + h * C + ch * 32;
#pragma unroll
    for (int q = 0; q < 8; ++q) {
        float4 v = srow[q];
        atomicAdd(&drow[q * 4 + 0], v.x * alpha);
        atomicAdd(&drow[q * 4 + 1], v.y * alpha);
        atomicAdd(&drow[q * 4 + 2], v.z * alpha);
        atomicAdd(&drow[q * 4 + 3], v.w * alpha);
    }
}

// bias (+ optional ReLU); F is a power of two
__global__ void k_bias_relu(float* __restrict__ p, const float* __restrict__ b, int n, int Fmask) {
    int i = blockIdx.x * blockDim.x + threadIdx.x;
    if (i >= n) return;
    float v = p[i] + b[i & Fmask];
    p[i] = v > 0.f ? v : 0.f;
}
__global__ void k_bias_add(float* __restrict__ p, const float* __restrict__ b, int n, int Fmask) {
    int i = blockIdx.x * blockDim.x + threadIdx.x;
    if (i >= n) return;
    p[i] += b[i & Fmask];
}

// ---------------- host side ----------------
extern "C" void kernel_launch(void* const* d_in, const int* in_sizes, int n_in,
                              void* d_out, int out_size, void* d_ws, size_t ws_size,
                              hipStream_t stream) {
    const float* x      = (const float*)d_in[0];
    const int*   ei     = (const int*)  d_in[1];   // [2, E] (src row, dst row)
    const float* W1     = (const float*)d_in[2];
    const float* a_src1 = (const float*)d_in[3];
    const float* a_dst1 = (const float*)d_in[4];
    const float* b1     = (const float*)d_in[5];
    const float* W2     = (const float*)d_in[6];
    const float* a_src2 = (const float*)d_in[7];
    const float* a_dst2 = (const float*)d_in[8];
    const float* b2     = (const float*)d_in[9];
    float* out = (float*)d_out;

    // ---- workspace layout (256B aligned) ----
    char* ws = (char*)d_ws;
    size_t off = 0;
    auto take = [&](size_t bytes) -> char* {
        char* p = ws + off;
        off = (off + bytes + 255) & ~(size_t)255;
        return p;
    };
    _Float16* W1p = (_Float16*)take((size_t)KDIM * F1 * 2);   // packed B fragments
    _Float16* W2p = (_Float16*)take((size_t)KDIM * F2 * 2);
    float*    hp1 = (float*)take((size_t)NNODES * F1 * 4);
    float*    as1 = (float*)take((size_t)NNODES * H1 * 4);
    float*    ad1 = (float*)take((size_t)NNODES * H1 * 4);
    unsigned* m1  = (unsigned*)take((size_t)NNODES * H1 * 4);
    float*    s1  = (float*)take((size_t)NNODES * H1 * 4);
    float*    eb  = (float*)take((size_t)NETOT * H1 * 4);     // reused by layer 2 (H=1)
    float*    ag1 = (float*)take((size_t)NNODES * F1 * 4);    // becomes x2 after bias+relu
    float*    hp2 = (float*)take((size_t)NNODES * F2 * 4);
    float*    as2 = (float*)take((size_t)NNODES * 4);
    float*    ad2 = (float*)take((size_t)NNODES * 4);
    unsigned* m2  = (unsigned*)take((size_t)NNODES * 4);
    float*    s2  = (float*)take((size_t)NNODES * 4);
    (void)ws_size; (void)n_in; (void)in_sizes; (void)out_size;

    const int T = 256;
    auto nb = [](int n, int t) { return (n + t - 1) / t; };

    // ---- zero accumulators (sortable(-inf) lower bound == 0 bits); all counts %4 == 0 ----
    k_zero_f4<<<nb(NNODES * H1 / 4, T), T, 0, stream>>>((float4*)m1, NNODES * H1 / 4);
    k_zero_f4<<<nb(NNODES * H1 / 4, T), T, 0, stream>>>((float4*)s1, NNODES * H1 / 4);
    k_zero_f4<<<nb(NNODES * F1 / 4, T), T, 0, stream>>>((float4*)ag1, NNODES * F1 / 4);
    k_zero_f4<<<nb(NNODES / 4, T), T, 0, stream>>>((float4*)m2, NNODES / 4);
    k_zero_f4<<<nb(NNODES / 4, T), T, 0, stream>>>((float4*)s2, NNODES / 4);
    k_zero_f4<<<nb(NNODES * F2 / 4, T), T, 0, stream>>>((float4*)out, NNODES * F2 / 4);

    // ---- pack weights into per-lane WMMA B-fragment images (f16) ----
    k_pack_B<<<nb(KDIM * F1, T), T, 0, stream>>>(W1, W1p, F1);
    k_pack_B<<<nb(KDIM * F2, T), T, 0, stream>>>(W2, W2p, F2);

    // ================= layer 1 (4 heads x 32) =================
    k_gemm_wmma<<<NNODES / 16, 256, 0, stream>>>(x, (const v16h*)W1p, hp1, F1);
    k_attn_terms<H1, C1><<<nb(NNODES * H1, T), T, 0, stream>>>(hp1, a_src1, a_dst1, as1, ad1);
    k_edge_max<H1><<<nb(NETOT * H1, T), T, 0, stream>>>(ei, as1, ad1, m1);
    k_edge_exp<H1><<<nb(NETOT * H1, T), T, 0, stream>>>(ei, as1, ad1, m1, eb, s1);
    k_edge_aggr<H1, C1><<<nb(NETOT * H1 * (C1 / 32), T), T, 0, stream>>>(ei, hp1, eb, s1, ag1);
    k_bias_relu<<<nb(NNODES * F1, T), T, 0, stream>>>(ag1, b1, NNODES * F1, F1 - 1);

    // ================= layer 2 (1 head x 256) =================
    k_gemm_wmma<<<NNODES / 16, 256, 0, stream>>>(ag1, (const v16h*)W2p, hp2, F2);
    k_attn_terms<1, F2><<<nb(NNODES, T), T, 0, stream>>>(hp2, a_src2, a_dst2, as2, ad2);
    k_edge_max<1><<<nb(NETOT, T), T, 0, stream>>>(ei, as2, ad2, m2);
    k_edge_exp<1><<<nb(NETOT, T), T, 0, stream>>>(ei, as2, ad2, m2, eb, s2);
    k_edge_aggr<1, F2><<<nb(NETOT * (F2 / 32), T), T, 0, stream>>>(ei, hp2, eb, s2, out);
    k_bias_add<<<nb(NNODES * F2, T), T, 0, stream>>>(out, b2, NNODES * F2, F2 - 1);
}
